// HierarchicalPLGBlock_12429635355033
// MI455X (gfx1250) — matrix operand order repaired and checked by reading, not compiled, not run on hardware
//
#include <hip/hip_runtime.h>
#include <hip/hip_bf16.h>

// ---------------------------------------------------------------------------
// HierarchicalPLGBlock for MI455X (gfx1250, wave32, WMMA).
//
// Pipeline:
//   1) cvt_norm: fp32 -> bf16 conversion + row |x|^2 norms (patches + labels,
//      and later the mood/genre outputs which become point sets).
//   2) dist_topk: WMMA bf16 16x16x32 dot-product tiles; ranking key
//      |p|^2 - 2*dot (label norm is row-constant, irrelevant for top-k);
//      per-wave register top-9 with unrolled sorted insertion; partial lists
//      per N-chunk written to scratch.
//   3) merge_topk: reduce per-chunk candidate lists to the final top-9 indices.
//   4) ctx: ctx[s][c] = max_j P[idx_j][c] - L[s][c]   (fp32 gather).
//   5) proj_ln: out = LayerNorm(L + concat(L, ctx...) @ W + b) * g + beta,
//      8 rows per block, concat cached in LDS, W streamed coalesced.
// ---------------------------------------------------------------------------

typedef __attribute__((ext_vector_type(16))) __bf16 v16bf;
typedef __attribute__((ext_vector_type(8)))  __bf16 v8bf;
typedef __attribute__((ext_vector_type(8)))  float  v8f;
typedef __attribute__((ext_vector_type(8)))  unsigned short u16x8;

#define PLG_C 256
#define PLG_K 9

__device__ __forceinline__ unsigned short plg_f32_to_bf16(float f) {
  unsigned int u = __float_as_uint(f);
  u += 0x7FFFu + ((u >> 16) & 1u);   // round-to-nearest-even
  return (unsigned short)(u >> 16);
}

__device__ __forceinline__ void plg_topk_insert(float (&kq)[PLG_K], int (&iq)[PLG_K],
                                                float key, int idx) {
  if (key < kq[PLG_K - 1]) {
    kq[PLG_K - 1] = key; iq[PLG_K - 1] = idx;
#pragma unroll
    for (int p = PLG_K - 1; p > 0; --p) {
      if (kq[p] < kq[p - 1]) {
        float tk = kq[p]; kq[p] = kq[p - 1]; kq[p - 1] = tk;
        int   ti = iq[p]; iq[p] = iq[p - 1]; iq[p - 1] = ti;
      }
    }
  }
}

// ---------------------------------------------------------------------------
// 1) fp32 -> bf16 + row norms. One wave per 256-wide row; 8 channels/lane.
// ---------------------------------------------------------------------------
__global__ __launch_bounds__(256) void plg_cvt_norm_kernel(
    const float* __restrict__ src, unsigned short* __restrict__ dst,
    float* __restrict__ norms, int rows) {
  int wv   = (int)((blockIdx.x * blockDim.x + threadIdx.x) >> 5);
  int lane = (int)(threadIdx.x & 31);
  if (wv >= rows) return;
  const float* r = src + (size_t)wv * PLG_C + lane * 8;
  float4 a = *(const float4*)(r);
  float4 b = *(const float4*)(r + 4);
  float acc = a.x * a.x + a.y * a.y + a.z * a.z + a.w * a.w
            + b.x * b.x + b.y * b.y + b.z * b.z + b.w * b.w;
  u16x8 o;
  o[0] = plg_f32_to_bf16(a.x); o[1] = plg_f32_to_bf16(a.y);
  o[2] = plg_f32_to_bf16(a.z); o[3] = plg_f32_to_bf16(a.w);
  o[4] = plg_f32_to_bf16(b.x); o[5] = plg_f32_to_bf16(b.y);
  o[6] = plg_f32_to_bf16(b.z); o[7] = plg_f32_to_bf16(b.w);
  *(u16x8*)(dst + (size_t)wv * PLG_C + lane * 8) = o;
#pragma unroll
  for (int off = 16; off > 0; off >>= 1) acc += __shfl_xor(acc, off, 32);
  if (lane == 0) norms[wv] = acc;
}

// ---------------------------------------------------------------------------
// 2) WMMA distance + per-chunk top-9.
//    One wave per (16-label tile, N-chunk). A (labels) resident in VGPRs,
//    B (patch rows) streamed; D = A*B^T accumulated over K=256 in 8 WMMAs.
// ---------------------------------------------------------------------------
__global__ __launch_bounds__(32) void plg_dist_topk_kernel(
    const unsigned short* __restrict__ Lbf,   // S x 256 bf16
    const unsigned short* __restrict__ Pbf,   // N x 256 bf16
    const float* __restrict__ Pnorm,          // N
    int n_tiles_total,                        // N/16
    int tiles_per_chunk,
    float* __restrict__ cand_key,             // [S][nchunks][9]
    int*   __restrict__ cand_idx) {
  const int lane = (int)threadIdx.x;
  const int m    = lane & 15;
  const int half = lane >> 4;
  const int tile = (int)blockIdx.x;
  const int chunk = (int)blockIdx.y;
  const int nchunks = (int)gridDim.y;

  // A fragments: 16x32 bf16 layout -> lane(m,half): K {k0+16h .. +15} contiguous
  v16bf afrag[8];
  {
    const unsigned short* lrow = Lbf + ((size_t)(tile * 16 + m)) * PLG_C;
#pragma unroll
    for (int t = 0; t < 8; ++t) {
      const int k0 = t * 32;
      v8bf lo = *(const v8bf*)(lrow + k0 + half * 16);
      v8bf hi = *(const v8bf*)(lrow + k0 + half * 16 + 8);
      afrag[t] = __builtin_shufflevector(lo, hi, 0, 1, 2, 3, 4, 5, 6, 7,
                                         8, 9, 10, 11, 12, 13, 14, 15);
    }
  }

  __shared__ float s_keys[16 * 16];
  __shared__ float s_mk[16 * 2 * PLG_K];
  __shared__ int   s_mi[16 * 2 * PLG_K];

  float kq[PLG_K]; int iq[PLG_K];
#pragma unroll
  for (int j = 0; j < PLG_K; ++j) { kq[j] = 3.0e38f; iq[j] = 0; }

  int t0 = chunk * tiles_per_chunk;
  int t1 = t0 + tiles_per_chunk;
  if (t1 > n_tiles_total) t1 = n_tiles_total;

  for (int pt = t0; pt < t1; ++pt) {
    const int colbase = pt * 16;
    // B fragment: lane(n=m, half): patch row (colbase+n), K {k0+16h .. +15} contiguous
    const unsigned short* prow = Pbf + ((size_t)(colbase + m)) * PLG_C + half * 16;
    if (pt + 1 < t1)  // CDNA5 global_prefetch of next patch tile
      __builtin_prefetch(Pbf + ((size_t)(colbase + 16 + m)) * PLG_C, 0, 1);

    v8f acc = {0.f, 0.f, 0.f, 0.f, 0.f, 0.f, 0.f, 0.f};
#pragma unroll
    for (int t = 0; t < 8; ++t) {
      v16bf bfrag = *(const v16bf*)(prow + t * 32);
      acc = __builtin_amdgcn_wmma_f32_16x16x32_bf16(
          false, afrag[t], false, bfrag, (short)0, acc, false, false);
    }
    // C/D layout: this lane holds column n=m, rows i + 8*half.
    const float pn = Pnorm[colbase + m];
#pragma unroll
    for (int i = 0; i < 8; ++i)
      s_keys[(i + 8 * half) * 16 + m] = pn - 2.0f * acc[i];
    __syncthreads();
    // Transposed scan: lane handles label row m, patch cols half*8..half*8+7.
#pragma unroll
    for (int j = 0; j < 8; ++j) {
      const int n = half * 8 + j;
      plg_topk_insert(kq, iq, s_keys[m * 16 + n], colbase + n);
    }
    __syncthreads();
  }

  // Merge the two half-wave lists per row.
#pragma unroll
  for (int j = 0; j < PLG_K; ++j) {
    s_mk[(m * 2 + half) * PLG_K + j] = kq[j];
    s_mi[(m * 2 + half) * PLG_K + j] = iq[j];
  }
  __syncthreads();
  if (half == 0) {
    float fk[PLG_K]; int fi[PLG_K];
#pragma unroll
    for (int j = 0; j < PLG_K; ++j) { fk[j] = 3.0e38f; fi[j] = 0; }
    for (int e = 0; e < 2 * PLG_K; ++e)
      plg_topk_insert(fk, fi, s_mk[m * 2 * PLG_K + e], s_mi[m * 2 * PLG_K + e]);
    const int srow = tile * 16 + m;
    float* ck = cand_key + ((size_t)srow * nchunks + chunk) * PLG_K;
    int*   ci = cand_idx + ((size_t)srow * nchunks + chunk) * PLG_K;
#pragma unroll
    for (int j = 0; j < PLG_K; ++j) { ck[j] = fk[j]; ci[j] = fi[j]; }
  }
}

// ---------------------------------------------------------------------------
// 3) Merge per-chunk candidates -> final sorted top-9 indices per row.
// ---------------------------------------------------------------------------
__global__ __launch_bounds__(32) void plg_merge_topk_kernel(
    const float* __restrict__ cand_key, const int* __restrict__ cand_idx,
    int nchunks, int* __restrict__ out_idx) {
  const int s = (int)blockIdx.x;
  const int lane = (int)threadIdx.x;
  const int total = nchunks * PLG_K;
  const float* ck = cand_key + (size_t)s * total;
  const int*   ci = cand_idx + (size_t)s * total;

  float kq[PLG_K]; int iq[PLG_K];
#pragma unroll
  for (int j = 0; j < PLG_K; ++j) { kq[j] = 3.0e38f; iq[j] = 0; }
  for (int e = lane; e < total; e += 32) plg_topk_insert(kq, iq, ck[e], ci[e]);

  __shared__ float sk[32 * PLG_K];
  __shared__ int   si[32 * PLG_K];
#pragma unroll
  for (int j = 0; j < PLG_K; ++j) { sk[lane * PLG_K + j] = kq[j]; si[lane * PLG_K + j] = iq[j]; }
  __syncthreads();
  if (lane == 0) {
    float fk[PLG_K]; int fi[PLG_K];
#pragma unroll
    for (int j = 0; j < PLG_K; ++j) { fk[j] = 3.0e38f; fi[j] = 0; }
    for (int e = 0; e < 32 * PLG_K; ++e) plg_topk_insert(fk, fi, sk[e], si[e]);
#pragma unroll
    for (int j = 0; j < PLG_K; ++j) out_idx[(size_t)s * PLG_K + j] = fi[j];
  }
}

// ---------------------------------------------------------------------------
// 4) ctx[s][c] = max_{j<kuse} P[idx_j][c] - L[s][c]   (fp32, exact)
// ---------------------------------------------------------------------------
__global__ __launch_bounds__(256) void plg_ctx_kernel(
    const float* __restrict__ L, const float* __restrict__ P,
    const int* __restrict__ topk, int kuse, float* __restrict__ ctx) {
  const int s = (int)blockIdx.x;
  const int c = (int)threadIdx.x;
  const int* ti = topk + (size_t)s * PLG_K;
  float mx = -3.0e38f;
  for (int j = 0; j < kuse; ++j) {
    const int idx = ti[j];
    mx = fmaxf(mx, P[(size_t)idx * PLG_C + c]);
  }
  ctx[(size_t)s * PLG_C + c] = mx - L[(size_t)s * PLG_C + c];
}

// ---------------------------------------------------------------------------
// 5) out = LN(L + concat(L, ctx0[, ctx1[, ctx2]]) @ W + b) * gamma + beta
//    8 rows / block; concat cached in LDS; thread c owns output channel c.
// ---------------------------------------------------------------------------
__global__ __launch_bounds__(256) void plg_proj_ln_kernel(
    const float* __restrict__ L,
    const float* __restrict__ ctx0, const float* __restrict__ ctx1,
    const float* __restrict__ ctx2, int nctx,
    const float* __restrict__ W, const float* __restrict__ bias,
    const float* __restrict__ gamma, const float* __restrict__ beta,
    float* __restrict__ out) {
  const int c = (int)threadIdx.x;
  const int row0 = (int)blockIdx.x * 8;
  const int inDim = (1 + nctx) * PLG_C;

  __shared__ float s_in[8 * 4 * PLG_C];   // up to 8 x 1024
  __shared__ float s_pre[8 * PLG_C];

  for (int r = 0; r < 8; ++r) {
    const size_t srow = (size_t)(row0 + r);
    s_in[r * inDim + c] = L[srow * PLG_C + c];
    if (nctx >= 1) s_in[r * inDim + 1 * PLG_C + c] = ctx0[srow * PLG_C + c];
    if (nctx >= 2) s_in[r * inDim + 2 * PLG_C + c] = ctx1[srow * PLG_C + c];
    if (nctx >= 3) s_in[r * inDim + 3 * PLG_C + c] = ctx2[srow * PLG_C + c];
  }
  __syncthreads();

  float acc[8];
#pragma unroll
  for (int r = 0; r < 8; ++r) acc[r] = bias[c];
  for (int i = 0; i < inDim; ++i) {
    const float w = W[(size_t)i * PLG_C + c];
#pragma unroll
    for (int r = 0; r < 8; ++r) acc[r] += s_in[r * inDim + i] * w;
  }
#pragma unroll
  for (int r = 0; r < 8; ++r) {
    acc[r] += L[(size_t)(row0 + r) * PLG_C + c];
    s_pre[r * PLG_C + c] = acc[r];
  }
  __syncthreads();

  // LayerNorm: wave w handles row w (8 waves, 8 channels/lane).
  const int wv = c >> 5, lane = c & 31;
  float x[8], sum = 0.f, sq = 0.f;
#pragma unroll
  for (int j = 0; j < 8; ++j) {
    x[j] = s_pre[wv * PLG_C + lane * 8 + j];
    sum += x[j]; sq += x[j] * x[j];
  }
#pragma unroll
  for (int off = 16; off > 0; off >>= 1) {
    sum += __shfl_xor(sum, off, 32);
    sq  += __shfl_xor(sq,  off, 32);
  }
  const float mu = sum * (1.0f / PLG_C);
  const float var = sq * (1.0f / PLG_C) - mu * mu;
  const float rstd = rsqrtf(var + 1e-5f);
  float* orow = out + (size_t)(row0 + wv) * PLG_C;
#pragma unroll
  for (int j = 0; j < 8; ++j) {
    const int ch = lane * 8 + j;
    orow[ch] = (x[j] - mu) * rstd * gamma[ch] + beta[ch];
  }
}

// ---------------------------------------------------------------------------
// Orchestration
// ---------------------------------------------------------------------------
static inline size_t plg_align(size_t x) { return (x + 255) & ~(size_t)255; }

extern "C" void kernel_launch(void* const* d_in, const int* in_sizes, int n_in,
                              void* d_out, int out_size, void* d_ws, size_t ws_size,
                              hipStream_t stream) {
  (void)n_in; (void)out_size; (void)ws_size;
  const int C = PLG_C;
  const float* patch = (const float*)d_in[0];
  const float* mood_emb  = (const float*)d_in[1];
  const float* genre_emb = (const float*)d_in[2];
  const float* sub_emb   = (const float*)d_in[3];
  const float* Wm = (const float*)d_in[4];  const float* bm = (const float*)d_in[5];
  const float* Wg = (const float*)d_in[6];  const float* bg = (const float*)d_in[7];
  const float* Ws = (const float*)d_in[8];  const float* bs = (const float*)d_in[9];
  const float* gm = (const float*)d_in[10]; const float* bnm = (const float*)d_in[11];
  const float* gg = (const float*)d_in[12]; const float* bng = (const float*)d_in[13];
  const float* gs = (const float*)d_in[14]; const float* bns = (const float*)d_in[15];

  const int N  = in_sizes[0] / C;                 // 100000 (divisible by 16)
  const int Sm = in_sizes[1] / C;                 // 64
  const int Sg = in_sizes[2] / C;                 // 256
  const int Ss = in_sizes[3] / C;                 // 1024
  const int Sall = Sm + Sg + Ss;                  // 1344
  const int NCHUNKS = 64;
  const int ntiles = N / 16;
  const int tpc = (ntiles + NCHUNKS - 1) / NCHUNKS;

  // workspace bump allocator
  char* ws = (char*)d_ws; size_t off = 0;
  auto alloc = [&](size_t bytes) { void* p = ws + off; off += plg_align(bytes); return p; };
  unsigned short* pbf   = (unsigned short*)alloc((size_t)N * C * 2);
  unsigned short* labbf = (unsigned short*)alloc((size_t)Sall * C * 2);
  unsigned short* mobf  = (unsigned short*)alloc((size_t)Sm * C * 2);
  unsigned short* gobf  = (unsigned short*)alloc((size_t)Sg * C * 2);
  float* pnorm  = (float*)alloc((size_t)N * 4);
  float* lnorm  = (float*)alloc((size_t)Sall * 4);   // written, unused for ranking
  float* monorm = (float*)alloc((size_t)Sm * 4);
  float* gonorm = (float*)alloc((size_t)Sg * 4);
  float* cand_key = (float*)alloc((size_t)Sall * NCHUNKS * PLG_K * 4);
  int*   cand_idx = (int*)  alloc((size_t)Sall * NCHUNKS * PLG_K * 4);
  int*   idx_all  = (int*)  alloc((size_t)Sall * PLG_K * 4);
  int*   idx_tmp  = (int*)  alloc((size_t)Ss * PLG_K * 4);
  float* ctx_m  = (float*)alloc((size_t)Sm * C * 4);
  float* ctx_gp = (float*)alloc((size_t)Sg * C * 4);
  float* ctx_gm = (float*)alloc((size_t)Sg * C * 4);
  float* ctx_sp = (float*)alloc((size_t)Ss * C * 4);
  float* ctx_sm = (float*)alloc((size_t)Ss * C * 4);
  float* ctx_sg = (float*)alloc((size_t)Ss * C * 4);

  float* out_mood  = (float*)d_out;
  float* out_genre = out_mood + (size_t)Sm * C;
  float* out_sub   = out_genre + (size_t)Sg * C;

  // 1) conversions + norms (patches and all label embeddings, labels concatenated)
  plg_cvt_norm_kernel<<<(N + 7) / 8, 256, 0, stream>>>(patch, pbf, pnorm, N);
  plg_cvt_norm_kernel<<<(Sm + 7) / 8, 256, 0, stream>>>(mood_emb, labbf, lnorm, Sm);
  plg_cvt_norm_kernel<<<(Sg + 7) / 8, 256, 0, stream>>>(genre_emb, labbf + (size_t)Sm * C, lnorm + Sm, Sg);
  plg_cvt_norm_kernel<<<(Ss + 7) / 8, 256, 0, stream>>>(sub_emb, labbf + (size_t)(Sm + Sg) * C, lnorm + Sm + Sg, Ss);

  // 2) all labels vs patches in one fused WMMA pass (patches streamed once)
  plg_dist_topk_kernel<<<dim3(Sall / 16, NCHUNKS), 32, 0, stream>>>(
      labbf, pbf, pnorm, ntiles, tpc, cand_key, cand_idx);
  plg_merge_topk_kernel<<<Sall, 32, 0, stream>>>(cand_key, cand_idx, NCHUNKS, idx_all);

  // 3) mood level
  plg_ctx_kernel<<<Sm, 256, 0, stream>>>(mood_emb, patch, idx_all, PLG_K, ctx_m);
  plg_proj_ln_kernel<<<Sm / 8, 256, 0, stream>>>(mood_emb, ctx_m, nullptr, nullptr, 1,
                                                 Wm, bm, gm, bnm, out_mood);
  plg_cvt_norm_kernel<<<(Sm + 7) / 8, 256, 0, stream>>>(out_mood, mobf, monorm, Sm);

  // 4) genre level (uses mood output as second point set)
  plg_dist_topk_kernel<<<dim3(Sg / 16, 1), 32, 0, stream>>>(
      labbf + (size_t)Sm * C, mobf, monorm, Sm / 16, Sm / 16, cand_key, cand_idx);
  plg_merge_topk_kernel<<<Sg, 32, 0, stream>>>(cand_key, cand_idx, 1, idx_tmp);
  plg_ctx_kernel<<<Sg, 256, 0, stream>>>(genre_emb, out_mood, idx_tmp, PLG_K / 2, ctx_gm);
  plg_ctx_kernel<<<Sg, 256, 0, stream>>>(genre_emb, patch, idx_all + (size_t)Sm * PLG_K, PLG_K, ctx_gp);
  plg_proj_ln_kernel<<<Sg / 8, 256, 0, stream>>>(genre_emb, ctx_gp, ctx_gm, nullptr, 2,
                                                 Wg, bg, gg, bng, out_genre);
  plg_cvt_norm_kernel<<<(Sg + 7) / 8, 256, 0, stream>>>(out_genre, gobf, gonorm, Sg);

  // 5) sub-genre level (mood + genre outputs as extra point sets)
  plg_dist_topk_kernel<<<dim3(Ss / 16, 1), 32, 0, stream>>>(
      labbf + (size_t)(Sm + Sg) * C, mobf, monorm, Sm / 16, Sm / 16, cand_key, cand_idx);
  plg_merge_topk_kernel<<<Ss, 32, 0, stream>>>(cand_key, cand_idx, 1, idx_tmp);
  plg_ctx_kernel<<<Ss, 256, 0, stream>>>(sub_emb, out_mood, idx_tmp, PLG_K / 3, ctx_sm);

  plg_dist_topk_kernel<<<dim3(Ss / 16, 1), 32, 0, stream>>>(
      labbf + (size_t)(Sm + Sg) * C, gobf, gonorm, Sg / 16, Sg / 16, cand_key, cand_idx);
  plg_merge_topk_kernel<<<Ss, 32, 0, stream>>>(cand_key, cand_idx, 1, idx_tmp);
  plg_ctx_kernel<<<Ss, 256, 0, stream>>>(sub_emb, out_genre, idx_tmp, PLG_K / 2, ctx_sg);

  plg_ctx_kernel<<<Ss, 256, 0, stream>>>(sub_emb, patch, idx_all + (size_t)(Sm + Sg) * PLG_K, PLG_K, ctx_sp);
  plg_proj_ln_kernel<<<Ss / 8, 256, 0, stream>>>(sub_emb, ctx_sp, ctx_sm, ctx_sg, 3,
                                                 Ws, bs, gs, bns, out_sub);
}
